// Forecaster_50689204027778
// MI455X (gfx1250) — matrix operand order repaired and checked by reading, not compiled, not run on hardware
//
#include <hip/hip_runtime.h>
#include <hip/hip_bf16.h>
#include <cstdint>
#include <cstddef>

typedef __attribute__((ext_vector_type(16))) _Float16 v16h;
typedef __attribute__((ext_vector_type(8)))  float    v8f;

#define LW 13   // number of flow fields

// ---------------------------------------------------------------------------
// Pack f32 NCHW activations into zero-padded, channel-interleaved f16:
//   dst[b][(c0+c)/16][(y+pad)*Wp + (x+pad)][(c0+c)%16]
// Each WMMA B-fragment becomes one contiguous 32B per-lane load.
// ---------------------------------------------------------------------------
__global__ void k_pack_act(const float* __restrict__ src, _Float16* __restrict__ dst,
                           int Bn, int C, int c0, int Ctot, int H, int W, int pad) {
  int i = blockIdx.x * blockDim.x + threadIdx.x;
  int total = Bn * C * H * W;
  if (i >= total) return;
  int x = i % W, y = (i / W) % H, c = (i / (W * H)) % C, b = i / (W * H * C);
  int Wp = W + 2 * pad;
  int plane = Wp * (H + 2 * pad);
  int cc = c0 + c;
  size_t di = ((size_t)(b * (Ctot >> 4) + (cc >> 4)) * plane +
               (size_t)(y + pad) * Wp + (x + pad)) * 16 + (cc & 15);
  dst[di] = (_Float16)src[i];
}

// ---------------------------------------------------------------------------
// Pack f32 conv weights into WMMA 16x32 f16 A-tile VGPR layout.
// Tile index = (tap*KB + kb)*OCB + ob ; within tile: lane*32B + half j.
// Packed A order: lanes 0-15 (M=lane): K = 0..15; lanes 16-31: K = 16..31 (K = 16*(lane/16) + j).
// flipT=1 folds in the stride-1 "deconv": w2[oc][ic][ky][kx] = w[ic][oc][2-ky][2-kx].
// ---------------------------------------------------------------------------
__global__ void k_pack_w(const float* __restrict__ w, _Float16* __restrict__ dst,
                         int OC, int IC, int k, int flipT) {
  int idx = blockIdx.x * blockDim.x + threadIdx.x;
  int KB = IC >> 5, OCB = (OC + 15) >> 4;
  int total = k * k * KB * OCB * 512;
  if (idx >= total) return;
  int j = idx & 15, l = (idx >> 4) & 31, tile = idx >> 9;
  int ob = tile % OCB, kb = (tile / OCB) % KB, tap = tile / (OCB * KB);
  int ky = tap / k, kx = tap % k;
  int m = l & 15, kg = l >> 4;
  int g = j >> 3, q = (j >> 1) & 3, sub = j & 1;
  int kk = kg * 16 + j;
  int ic = kb * 32 + kk;
  int oc = ob * 16 + m;
  float v = 0.f;
  if (oc < OC) {
    if (flipT) v = w[((size_t)ic * OC + oc) * (k * k) + (k - 1 - ky) * k + (k - 1 - kx)];
    else       v = w[((size_t)oc * IC + ic) * (k * k) + ky * k + kx];
  }
  dst[idx] = (_Float16)v;
}

// ---------------------------------------------------------------------------
// Implicit-GEMM conv: k*k shifted 1x1 GEMMs with v_wmma_f32_16x16x32_f16.
// Block = 4 waves, one OC-block; A tiles for each tap are staged in LDS once
// per block (all waves share them). Each wave owns a 16(OC) x 32(pixel) tile
// (two accumulators -> 2 WMMA per A fragment).
// Grid.x = H*W/128 exactly (all sizes divide), so no guard: EXEC all-ones.
// mode: 0 = none, 1 = leaky(0.2), 2 = sigmoid.
// ---------------------------------------------------------------------------
__global__ __launch_bounds__(128)
void k_conv_wmma(const _Float16* __restrict__ act, const _Float16* __restrict__ wgt,
                 const float* __restrict__ bias, float* __restrict__ out,
                 int C16, int KB, int OC, int H, int Wshift, int k, int pad, int mode) {
  __shared__ __align__(16) _Float16 smem[21504];   // up to KB=42 tiles of 1KB
  const int W = 1 << Wshift;
  const int lane = threadIdx.x & 31;
  const int wv   = threadIdx.x >> 5;
  const int tid  = threadIdx.x;
  const int ob = blockIdx.y, b = blockIdx.z, OCB = gridDim.y;
  const int Wp = W + 2 * pad;
  const int plane = Wp * (H + 2 * pad);
  const int n = lane & 15, kg = lane >> 4;
  const int p0 = (blockIdx.x * 4 + wv) << 5;       // 32 consecutive pixels
  const int p1 = p0 + 16;
  const int Wm1 = W - 1;
  const int y0 = p0 >> Wshift, x0 = p0 & Wm1;
  const int y1 = p1 >> Wshift, x1 = p1 & Wm1;
  const _Float16* act_b = act + (size_t)b * C16 * plane * 16;
  const _Float16* a0 = act_b + (((size_t)y0 * Wp + x0 + n) << 4);
  const _Float16* a1 = act_b + (((size_t)y1 * Wp + x1 + n) << 4);
  const uint4* wsrc = (const uint4*)wgt;
  v8f acc0 = {}, acc1 = {};
  const size_t cstep = (size_t)plane << 5;         // halves per K-block (2 c16 planes)
  int tap = 0;
  for (int ky = 0; ky < k; ++ky) {
    for (int kx = 0; kx < k; ++kx, ++tap) {
      __syncthreads();                              // previous tap fully consumed
      {
        uint4* sm4 = (uint4*)smem;
        const int chunks = KB << 6;                 // 64 x uint4 per 1KB tile
        const size_t base = (size_t)(tap * KB) * OCB + ob;
        for (int c = tid; c < chunks; c += 128) {
          int kb = c >> 6, wi = c & 63;
          sm4[c] = wsrc[(base + (size_t)kb * OCB) * 64 + wi];
        }
        if (tap + 1 < k * k)                        // global_prefetch_b8 of next tap
          __builtin_prefetch(wsrc + ((size_t)((tap + 1) * KB) * OCB + ob) * 64, 0, 1);
      }
      __syncthreads();
      const _Float16* c0 = a0 + (((size_t)ky * Wp + kx) << 4) + ((size_t)kg * plane << 4);
      const _Float16* c1 = a1 + (((size_t)ky * Wp + kx) << 4) + ((size_t)kg * plane << 4);
      const _Float16* sa = smem + (lane << 4);
#pragma unroll 2
      for (int kb = 0; kb < KB; ++kb) {
        v16h a  = *(const v16h*)(sa + (kb << 9));   // ds_load, shared across waves
        v16h b0 = *(const v16h*)(c0);
        v16h b1 = *(const v16h*)(c1);
        c0 += cstep; c1 += cstep;
        acc0 = __builtin_amdgcn_wmma_f32_16x16x32_f16(
            false, a, false, b0, (short)0, acc0, false, false);
        acc1 = __builtin_amdgcn_wmma_f32_16x16x32_f16(
            false, a, false, b1, (short)0, acc1, false, false);
      }
    }
  }
  const int HW = H << Wshift;
  float* outb = out + (size_t)b * OC * HW;
  for (int r = 0; r < 8; ++r) {
    int m  = r + (kg << 3);                         // C/D layout: lanes>=16 -> M = r+8
    int oc = (ob << 4) + m;
    if (oc < OC) {
      float bi = bias[oc];
      float v0 = acc0[r] + bi, v1 = acc1[r] + bi;
      if (mode == 1) {
        v0 = (v0 >= 0.f) ? v0 : 0.2f * v0;
        v1 = (v1 >= 0.f) ? v1 : 0.2f * v1;
      } else if (mode == 2) {
        v0 = 1.f / (1.f + __expf(-v0));
        v1 = 1.f / (1.f + __expf(-v1));
      }
      outb[(size_t)oc * HW + p0 + n] = v0;
      outb[(size_t)oc * HW + p1 + n] = v1;
    }
  }
}

// ---------------------------------------------------------------------------
// 13-way flow warp: bilinear sample of prev at grid + (u_l, v_l).
// ---------------------------------------------------------------------------
__global__ void k_warp(const float* __restrict__ prev, const float* __restrict__ flow,
                       float* __restrict__ M, int Bn, int C, int H, int W) {
  int i = blockIdx.x * blockDim.x + threadIdx.x;
  int HW = H * W;
  int total = Bn * LW * C * HW;
  if (i >= total) return;
  int x = i % W, y = (i / W) % H;
  int c = (i / HW) % C;
  int l = (i / (HW * C)) % LW;
  int b = i / (HW * C * LW);
  float u = flow[(size_t)(b * 2 * LW + l) * HW + y * W + x];
  float v = flow[(size_t)(b * 2 * LW + LW + l) * HW + y * W + x];
  float bx = (float)x * 2.f / W - 1.f;
  float by = (float)y * 2.f / W - 1.f;   // reference divides by W for both axes
  float ix = (((bx + u) + 1.f) * W - 1.f) * 0.5f;
  float iy = (((by + v) + 1.f) * H - 1.f) * 0.5f;
  float xf = floorf(ix), yf = floorf(iy);
  float wx = ix - xf, wy = iy - yf;
  int x0 = (int)xf, y0 = (int)yf;
  const float* im = prev + (size_t)(b * C + c) * HW;
  auto g = [&](int yy, int xx) -> float {
    return (yy >= 0 && yy < H && xx >= 0 && xx < W) ? im[yy * W + xx] : 0.f;
  };
  M[i] = g(y0, x0)         * (1.f - wy) * (1.f - wx)
       + g(y0, x0 + 1)     * (1.f - wy) * wx
       + g(y0 + 1, x0)     * wy         * (1.f - wx)
       + g(y0 + 1, x0 + 1) * wy         * wx;
}

// M *= broadcast(r) over the L axis (channel l*C+c uses r[c])
__global__ void k_mul_reset(float* __restrict__ M, const float* __restrict__ r,
                            int Bn, int C, int HW) {
  int i = blockIdx.x * blockDim.x + threadIdx.x;
  int total = Bn * LW * C * HW;
  if (i >= total) return;
  int p = i % HW;
  int c = (i / HW) % C;
  int b = i / (HW * C * LW);
  M[i] *= r[(size_t)(b * C + c) * HW + p];
}

// s = prev*(1-u) + o*u
__global__ void k_combine(const float* __restrict__ prev, const float* __restrict__ u,
                          const float* __restrict__ o, float* __restrict__ s, int n) {
  int i = blockIdx.x * blockDim.x + threadIdx.x;
  if (i >= n) return;
  float uu = u[i];
  s[i] = prev[i] * (1.f - uu) + o[i] * uu;
}

// ---------------------------------------------------------------------------
// Stride-2 4x4 transposed conv + leaky; exactly 2x2 taps per output pixel.
// w layout (IC, OC, 4, 4).
// ---------------------------------------------------------------------------
__global__ void k_deconv_up(const float* __restrict__ s, const float* __restrict__ w,
                            const float* __restrict__ bias, float* __restrict__ out,
                            int Bn, int IC, int OC, int H, int W) {
  int H2 = H * 2, W2 = W * 2;
  int i = blockIdx.x * blockDim.x + threadIdx.x;
  int total = Bn * OC * H2 * W2;
  if (i >= total) return;
  int x2 = i % W2, y2 = (i / W2) % H2;
  int oc = (i / (W2 * H2)) % OC;
  int b  = i / (W2 * H2 * OC);
  int py = (y2 + 1) & 1, px = (x2 + 1) & 1;
  int iy0 = (y2 + 1 - py) >> 1, ix0 = (x2 + 1 - px) >> 1;
  float acc = bias[oc];
  for (int t = 0; t < 4; ++t) {
    int dy = t >> 1, dx = t & 1;
    int ky = py + dy * 2, kx = px + dx * 2;
    int iy = iy0 - dy, ix = ix0 - dx;
    if (iy < 0 || iy >= H || ix < 0 || ix >= W) continue;
    const float* ip = s + (size_t)(b * IC) * H * W + (size_t)iy * W + ix;
    const float* wp = w + (size_t)oc * 16 + ky * 4 + kx;
    for (int ic = 0; ic < IC; ++ic)
      acc += ip[(size_t)ic * H * W] * wp[(size_t)ic * OC * 16];
  }
  out[i] = (acc >= 0.f) ? acc : 0.2f * acc;
}

// ===========================================================================
// Host orchestration
// ===========================================================================
#define MBy ((size_t)1 << 20)

extern "C" void kernel_launch(void* const* d_in, const int* in_sizes, int n_in,
                              void* d_out, int out_size, void* d_ws, size_t ws_size,
                              hipStream_t stream) {
  (void)in_sizes; (void)n_in; (void)out_size; (void)ws_size;
  const int Bn = 4;
  const int CH[3]   = {96, 96, 64};
  const int CIN_[3] = {0, 96, 96};
  const int HH[3]   = {16, 32, 64};

  // Input flattening (jax pytree, dict keys alphabetical):
  // 0:h0 1:h1 2:h2, cells[i] x3: {bo,br,bu,db1,db2,dw1,dw2,wo,wr,wu},
  // 33:out_b 34:out_w 35-37:up_b[0..2] 38-40:up_w[0..2]
  auto FIN = [&](int i) -> const float* { return (const float*)d_in[i]; };
  const float* hst[3] = {FIN(0), FIN(1), FIN(2)};

  uint8_t* ws = (uint8_t*)d_ws;
  float*     X[2] = {(float*)(ws + 0 * MBy), (float*)(ws + 8 * MBy)};
  _Float16*  P    = (_Float16*)(ws + 16 * MBy);   // packed padded activations (36MB)
  float*     T1   = (float*)(ws + 52 * MBy);      // conv1 out (3MB)
  float*     T2   = (float*)(ws + 55 * MBy);      // flow (3MB)
  float*     Mb   = (float*)(ws + 58 * MBy);      // warped M (56MB)
  float*     Rb   = (float*)(ws + 114 * MBy);
  float*     Ub   = (float*)(ws + 119 * MBy);
  float*     Ob   = (float*)(ws + 124 * MBy);
  float*     Sb   = (float*)(ws + 129 * MBy);
  _Float16*  WP   = (_Float16*)(ws + 134 * MBy);  // packed weights (4MB)

  auto packA = [&](const float* src, int C, int c0, int Ctot, int H, int W, int pad) {
    int total = Bn * C * H * W;
    k_pack_act<<<dim3((total + 255) / 256), dim3(256), 0, stream>>>(
        src, P, Bn, C, c0, Ctot, H, W, pad);
  };
  auto packW = [&](const float* w, int OC, int IC, int k, int flipT) {
    int total = k * k * (IC / 32) * ((OC + 15) / 16) * 512;
    k_pack_w<<<dim3((total + 255) / 256), dim3(256), 0, stream>>>(w, WP, OC, IC, k, flipT);
  };
  auto conv = [&](const float* bias, float* out, int C, int OC,
                  int H, int W, int k, int pad, int mode) {
    int OCB = (OC + 15) / 16;
    int wshift = __builtin_ctz(W);
    dim3 grid((H * W) / 128, OCB, Bn);              // exact for all our sizes
    k_conv_wmma<<<grid, dim3(128), 0, stream>>>(
        P, WP, bias, out, C / 16, C / 32, OC, H, wshift, k, pad, mode);
  };

  const float* x = nullptr;
  int cur = 0;
  for (int lvl = 0; lvl < 3; ++lvl) {
    const int H = HH[lvl], W = HH[lvl], ch = CH[lvl], cin = CIN_[lvl];
    const int cg = cin + ch * LW;
    const int HW = H * W;
    const float* prev = hst[lvl];
    const int ib = 3 + lvl * 10;
    const float *bo = FIN(ib + 0), *br = FIN(ib + 1), *bu = FIN(ib + 2);
    const float *db1 = FIN(ib + 3), *db2 = FIN(ib + 4);
    const float *dw1 = FIN(ib + 5), *dw2 = FIN(ib + 6);
    const float *wo = FIN(ib + 7), *wr = FIN(ib + 8), *wu = FIN(ib + 9);

    // --- conv1 (5x5, OC=32, leaky) on concat(x, prev) ---
    const int Cs = cin + ch;
    hipMemsetAsync(P, 0, (size_t)Bn * Cs * (H + 4) * (W + 4) * 2, stream);
    if (x) packA(x, cin, 0, Cs, H, W, 2);
    packA(prev, ch, cin, Cs, H, W, 2);
    packW(dw1, 32, Cs, 5, 0);
    conv(db1, T1, Cs, 32, H, W, 5, 2, 1);

    // --- conv2 (5x5, OC=26, leaky) ---
    hipMemsetAsync(P, 0, (size_t)Bn * 32 * (H + 4) * (W + 4) * 2, stream);
    packA(T1, 32, 0, 32, H, W, 2);
    packW(dw2, 26, 32, 5, 0);
    conv(db2, T2, 32, 26, H, W, 5, 2, 1);

    // --- 13-way bilinear warp of prev ---
    int mtot = Bn * LW * ch * HW;
    k_warp<<<dim3((mtot + 255) / 256), dim3(256), 0, stream>>>(prev, T2, Mb, Bn, ch, H, W);

    // --- pack stack2 = concat(x, M), pad 1 ---
    hipMemsetAsync(P, 0, (size_t)Bn * cg * (H + 2) * (W + 2) * 2, stream);
    if (x) packA(x, cin, 0, cg, H, W, 1);
    packA(Mb, ch * LW, cin, cg, H, W, 1);

    // --- r = sigmoid(conv3x3) ; u = sigmoid(conv3x3) ---
    packW(wr, ch, cg, 3, 1);
    conv(br, Rb, cg, ch, H, W, 3, 1, 2);
    packW(wu, ch, cg, 3, 1);
    conv(bu, Ub, cg, ch, H, W, 3, 1, 2);

    // --- M *= broadcast(r); repack M portion; o = leaky(conv3x3) ---
    k_mul_reset<<<dim3((mtot + 255) / 256), dim3(256), 0, stream>>>(Mb, Rb, Bn, ch, HW);
    packA(Mb, ch * LW, cin, cg, H, W, 1);
    packW(wo, ch, cg, 3, 1);
    conv(bo, Ob, cg, ch, H, W, 3, 1, 1);

    // --- GRU combine: s = prev*(1-u) + o*u ---
    int stot = Bn * ch * HW;
    k_combine<<<dim3((stot + 255) / 256), dim3(256), 0, stream>>>(prev, Ub, Ob, Sb, stot);

    // --- x_next = leaky(deconv 4x4 stride2 pad1) ---
    float* xn = X[cur]; cur ^= 1;
    int utot = Bn * ch * (2 * H) * (2 * W);
    k_deconv_up<<<dim3((utot + 255) / 256), dim3(256), 0, stream>>>(
        Sb, FIN(38 + lvl), FIN(35 + lvl), xn, Bn, ch, ch, H, W);
    x = xn;
  }

  // --- final 1x1 conv: (4,64,128,128) -> (4,1,128,128) ---
  hipMemsetAsync(P, 0, (size_t)Bn * 64 * 128 * 128 * 2, stream);
  packA(x, 64, 0, 64, 128, 128, 0);
  packW(FIN(34), 1, 64, 1, 0);
  conv(FIN(33), (float*)d_out, 64, 1, 128, 128, 1, 0, 0);
}
